// ESCBlock_1812476199068
// MI455X (gfx1250) — matrix operand order repaired and checked by reading, not compiled, not run on hardware
//
#include <hip/hip_runtime.h>
#include <math.h>

// ---------------------------------------------------------------------------
// CDNA5 WMMA types
// ---------------------------------------------------------------------------
typedef __attribute__((ext_vector_type(16))) _Float16 v16h;
typedef __attribute__((ext_vector_type(2)))  _Float16 v2h;
typedef __attribute__((ext_vector_type(8)))  float    v8f;

__device__ __forceinline__ float geluf(float x) {
    return 0.5f * x * (1.0f + erff(x * 0.70710678118654752f));
}

// ---------------------------------------------------------------------------
// Generic 1x1-conv GEMM via v_wmma_f32_16x16x32_f16.
// out[b, co, p] = act( sum_ci w[co,ci] * in[b,ci,p] + bias[co] )
// 256 threads = 8 waves; tile M=64 (co) x N=64 (px); K staged 32/iter.
//
// Fragments live in LDS already in WMMA lane order (ISA 7.12.2):
//   A (16x32, m,k): lane = ((k&8)<<1)+(m&15), pos = (k&7)|((k&16)>>1)
//   B (32x16, k,n): lane = ((k&8)<<1)+(n&15), pos = (k&7)|((k&16)>>1)
// -> each lane reads its fragment as one contiguous 32B v16h (ds_load_b128 x2).
//
// Staging is branchless (clamped indices + select, 32-bit addressing) and
// double-buffered: next tile's 16 global loads issue before the WMMAs,
// cvt+ds_store after, one barrier per K step.
// ---------------------------------------------------------------------------
__device__ __forceinline__ void stage_load(const float* __restrict__ in_b,
                                           const float* __restrict__ wrow,
                                           int Cin, int HW, int pc, int kb, int k0,
                                           float xv[8], float wv[8])
{
#pragma unroll
    for (int e = 0; e < 4; ++e) {
        int k   = kb + 8 * e;
        int kk0 = k0 + k;
        int kc0 = kk0 < Cin - 1 ? kk0 : Cin - 1;
        int kc1 = kk0 + 1 < Cin - 1 ? kk0 + 1 : Cin - 1;
        xv[2 * e]     = in_b[kc0 * HW + pc];
        xv[2 * e + 1] = in_b[kc1 * HW + pc];
        wv[2 * e]     = wrow[kc0];
        wv[2 * e + 1] = wrow[kc1];
    }
}

__device__ __forceinline__ void stage_store(v16h (*Af)[32], v16h (*Bf)[32],
                                            int Cin, int k0, int kb,
                                            int n, int m, bool mok, bool pok,
                                            const float xv[8], const float wv[8])
{
#pragma unroll
    for (int e = 0; e < 4; ++e) {
        int k   = kb + 8 * e;
        bool ok0 = pok && (k0 + k     < Cin);
        bool ok1 = pok && (k0 + k + 1 < Cin);
        float x0 = ok0 ? xv[2 * e]     : 0.0f;
        float x1 = ok1 ? xv[2 * e + 1] : 0.0f;
        v2h px = { (_Float16)x0, (_Float16)x1 };
        int lnB  = ((k & 8) << 1) + (n & 15);
        int posB = (k & 7) | ((k & 16) >> 1);
        *(v2h*)&(((_Float16*)&Bf[n >> 4][lnB])[posB]) = px;

        bool wk0 = mok && (k0 + k     < Cin);
        bool wk1 = mok && (k0 + k + 1 < Cin);
        float w0 = wk0 ? wv[2 * e]     : 0.0f;
        float w1 = wk1 ? wv[2 * e + 1] : 0.0f;
        v2h pw = { (_Float16)w0, (_Float16)w1 };
        int lnA  = ((k & 8) << 1) + (m & 15);
        int posA = (k & 7) | ((k & 16) >> 1);
        *(v2h*)&(((_Float16*)&Af[m >> 4][lnA])[posA]) = pw;
    }
}

__global__ void gemm1x1_wmma_k(const float* __restrict__ in,
                               const float* __restrict__ wgt,
                               const float* __restrict__ bias,
                               float* __restrict__ out,
                               int Cin, int Cout, int HW, int act)
{
    __shared__ v16h Af[2][4][32];   // [buf][co subtile][lane]  : 8 KB
    __shared__ v16h Bf[2][4][32];   // [buf][16-px group][lane] : 8 KB

    const int tid  = threadIdx.x;
    const int lane = tid & 31;
    const int wave = tid >> 5;
    const int lm   = lane & 15;
    const int hf   = lane >> 4;
    const int wm   = wave & 3;
    const int wn   = wave >> 2;

    const int b  = blockIdx.z;
    const int p0 = blockIdx.x * 64;
    const int m0 = blockIdx.y * 64;

    const float* in_b  = in  + (size_t)b * Cin  * HW;
    float*       out_b = out + (size_t)b * Cout * HW;

    // per-thread staging geometry (loop-invariant)
    const int n  = tid & 63;                 // pixel column within tile / weight row
    const int m  = n;
    const int kb = (tid >> 6) << 1;          // base k slot (0,2,4,6)
    const int p  = p0 + n;
    const bool pok = p < HW;
    const int pc = pok ? p : HW - 1;
    const int co_stage = m0 + m;
    const bool mok = co_stage < Cout;
    const float* wrow = wgt + (size_t)(mok ? co_stage : Cout - 1) * Cin;

    v8f acc0 = {};
    v8f acc1 = {};

    const int nk = (Cin + 31) >> 5;
    float xv[8], wv[8];

    // prologue: stage tile 0
    stage_load(in_b, wrow, Cin, HW, pc, kb, 0, xv, wv);
    stage_store(Af[0], Bf[0], Cin, 0, kb, n, m, mok, pok, xv, wv);
    __syncthreads();

    for (int t = 0; t < nk; ++t) {
        const int cur = t & 1;
        const bool more = (t + 1) < nk;
        if (more)   // uniform branch
            stage_load(in_b, wrow, Cin, HW, pc, kb, (t + 1) * 32, xv, wv);

        v16h a  = Af[cur][wm][lane];         // ds_load_b128 x2
        v16h b0 = Bf[cur][wn * 2][lane];
        v16h b1 = Bf[cur][wn * 2 + 1][lane];

        acc0 = __builtin_amdgcn_wmma_f32_16x16x32_f16(false, a, false, b0,
                                                      (short)0, acc0, false, false);
        acc1 = __builtin_amdgcn_wmma_f32_16x16x32_f16(false, a, false, b1,
                                                      (short)0, acc1, false, false);

        if (more)
            stage_store(Af[1 - cur], Bf[1 - cur], Cin, (t + 1) * 32,
                        kb, n, m, mok, pok, xv, wv);
        __syncthreads();
    }

    // C/D layout: VGPR j -> M = j + 8*hf ; N = lane&15
#pragma unroll
    for (int j = 0; j < 8; ++j) {
        int mm = j + 8 * hf;
        int co = m0 + wm * 16 + mm;
        if (co >= Cout) continue;
        float bv = bias[co];
        int pp = p0 + wn * 32 + lm;
        if (pp < HW) {
            float v = acc0[j] + bv;
            if (act) v = geluf(v);
            out_b[(size_t)co * HW + pp] = v;
        }
        int p2 = pp + 16;
        if (p2 < HW) {
            float v = acc1[j] + bv;
            if (act) v = geluf(v);
            out_b[(size_t)co * HW + p2] = v;
        }
    }
}

// ---------------------------------------------------------------------------
// Channel-first layernorm over C at each pixel (two-pass; L2-resident).
// ---------------------------------------------------------------------------
__global__ void layernorm_cf_k(const float* __restrict__ x,
                               const float* __restrict__ w,
                               const float* __restrict__ bvec,
                               float* __restrict__ out,
                               int C, int HW, int total)
{
    int idx = blockIdx.x * blockDim.x + threadIdx.x;
    if (idx >= total) return;
    int b = idx / HW, p = idx - b * HW;
    const float* xb = x + (size_t)b * C * HW + p;
    float s = 0.0f, s2 = 0.0f;
    for (int c = 0; c < C; ++c) {
        float v = xb[(size_t)c * HW];
        s += v; s2 += v * v;
    }
    float mu  = s / (float)C;
    float var = s2 / (float)C - mu * mu;
    float inv = rsqrtf(var + 1e-6f);
    float* ob = out + (size_t)b * C * HW + p;
    for (int c = 0; c < C; ++c) {
        float v = xb[(size_t)c * HW];
        ob[(size_t)c * HW] = (v - mu) * inv * w[c] + bvec[c];
    }
}

// ---------------------------------------------------------------------------
// Depthwise 3x3 (pad 1) + GELU + residual:  out = gelu(dw(in)) + in
// ---------------------------------------------------------------------------
__global__ void dw3x3_gelu_add_k(const float* __restrict__ in,
                                 const float* __restrict__ w,
                                 const float* __restrict__ bvec,
                                 float* __restrict__ out,
                                 int C, int H, int W, int total)
{
    int idx = blockIdx.x * blockDim.x + threadIdx.x;
    if (idx >= total) return;
    int HW = H * W;
    int p  = idx % HW;
    int bc = idx / HW;
    int c  = bc % C;
    int hh = p / W, ww = p - hh * W;
    const float* plane = in + (size_t)bc * HW;
    float s = bvec[c];
    for (int u = 0; u < 3; ++u) {
        int y = hh + u - 1;
        if (y < 0 || y >= H) continue;
        for (int v = 0; v < 3; ++v) {
            int xw = ww + v - 1;
            if (xw < 0 || xw >= W) continue;
            s += w[c * 9 + u * 3 + v] * plane[(size_t)y * W + xw];
        }
    }
    out[idx] = geluf(s) + in[idx];
}

__global__ void add_inplace_k(float* __restrict__ dst, const float* __restrict__ src, int n)
{
    int idx = blockIdx.x * blockDim.x + threadIdx.x;
    if (idx < n) dst[idx] += src[idx];
}

// copy 16-ch compact buffer into first 16 channels of a 64-ch buffer
__global__ void copy16_k(const float* __restrict__ src, float* __restrict__ dst,
                         int HW, int n /* B*16*HW */)
{
    int idx = blockIdx.x * blockDim.x + threadIdx.x;
    if (idx >= n) return;
    int p = idx % HW;
    int bc = idx / HW;            // b*16 + c
    int b = bc >> 4, c = bc & 15;
    dst[((size_t)b * 64 + c) * HW + p] = src[idx];
}

// ---------------------------------------------------------------------------
// Geometric ensemble of the 13x13 PLK filter (16x16x13x13).
// kr[a,b] = k[12-b, a]  (rot90 k=-1)
// ---------------------------------------------------------------------------
__global__ void geo_ensemble_k(const float* __restrict__ k, float* __restrict__ out, int n)
{
    int idx = blockIdx.x * blockDim.x + threadIdx.x;
    if (idx >= n) return;
    int r  = idx % 169;
    int ij = idx / 169;
    int a = r / 13, b = r - a * 13;
    const float* kk = k + (size_t)ij * 169;
#define KAT(u, v) kk[(u) * 13 + (v)]
    float s = KAT(a, b) + KAT(a, 12 - b) + KAT(12 - a, b) + KAT(12 - a, 12 - b)
            + KAT(12 - b, a) + KAT(b, a) + KAT(12 - b, 12 - a) + KAT(b, 12 - a);
#undef KAT
    out[idx] = s * 0.125f;
}

// ---------------------------------------------------------------------------
// Global mean over H,W of first 16 channels of a 64-ch tensor.
// grid = B*16 blocks, 256 threads.
// ---------------------------------------------------------------------------
__global__ void gmean16_k(const float* __restrict__ x, float* __restrict__ g, int HW)
{
    __shared__ float sdata[256];
    int bc = blockIdx.x;              // b*16 + c
    int b = bc >> 4, c = bc & 15;
    const float* plane = x + ((size_t)b * 64 + c) * HW;
    float s = 0.0f;
    for (int p = threadIdx.x; p < HW; p += 256) s += plane[p];
    sdata[threadIdx.x] = s;
    __syncthreads();
    for (int off = 128; off > 0; off >>= 1) {
        if (threadIdx.x < off) sdata[threadIdx.x] += sdata[threadIdx.x + off];
        __syncthreads();
    }
    if (threadIdx.x == 0) g[bc] = sdata[0] / (float)HW;
}

// ---------------------------------------------------------------------------
// Dynamic kernel MLP: h1 = gelu(W1 g + b1) (16->8), dynk = W2 h1 + b2 (8->144)
// grid = B, block 256.
// ---------------------------------------------------------------------------
__global__ void dyn_kernels_k(const float* __restrict__ g,
                              const float* __restrict__ w1, const float* __restrict__ b1,
                              const float* __restrict__ w2, const float* __restrict__ b2,
                              float* __restrict__ dynk)
{
    __shared__ float h1s[8];
    int b = blockIdx.x;
    if (threadIdx.x < 8) {
        float s = b1[threadIdx.x];
        for (int k = 0; k < 16; ++k) s += w1[threadIdx.x * 16 + k] * g[b * 16 + k];
        h1s[threadIdx.x] = geluf(s);
    }
    __syncthreads();
    if (threadIdx.x < 144) {
        float s = b2[threadIdx.x];
        for (int k = 0; k < 8; ++k) s += w2[threadIdx.x * 8 + k] * h1s[k];
        dynk[b * 144 + threadIdx.x] = s;
    }
}

// ---------------------------------------------------------------------------
// conv_attention core: out16 = conv13x13(x1, lk, pad6) + dyn_dw3x3(x1, dynk)
// x1 = first 16 channels of the 64-ch y buffer. 16x16 spatial tile per block,
// 28x28x16 input patch in LDS; lk reads are thread-uniform (scalarized, L2).
// ---------------------------------------------------------------------------
__global__ void conv_attn16_k(const float* __restrict__ y,
                              const float* __restrict__ lk,
                              const float* __restrict__ dynk,
                              float* __restrict__ out16,
                              int H, int W)
{
    __shared__ float patch[16][28][28];   // ~50 KB
    const int HW = H * W;
    const int b  = blockIdx.z;
    const int h0 = blockIdx.y * 16 - 6;
    const int w0 = blockIdx.x * 16 - 6;

    for (int idx = threadIdx.x; idx < 16 * 28 * 28; idx += 256) {
        int ci = idx / 784;
        int r  = idx - ci * 784;
        int i = r / 28, j = r - i * 28;
        int yy = h0 + i, xx = w0 + j;
        float v = 0.0f;
        if (yy >= 0 && yy < H && xx >= 0 && xx < W)
            v = y[((size_t)b * 64 + ci) * HW + (size_t)yy * W + xx];
        patch[ci][i][j] = v;
    }
    __syncthreads();

    int i = threadIdx.x >> 4, j = threadIdx.x & 15;
    int h = blockIdx.y * 16 + i, w = blockIdx.x * 16 + j;
    if (h >= H || w >= W) return;

    for (int co = 0; co < 16; ++co) {
        float s = 0.0f;
        for (int ci = 0; ci < 16; ++ci) {
            const float* lkp = lk + ((size_t)co * 16 + ci) * 169;
            for (int u = 0; u < 13; ++u)
                for (int v = 0; v < 13; ++v)
                    s += lkp[u * 13 + v] * patch[ci][i + u][j + v];
        }
        const float* dk = dynk + ((size_t)b * 16 + co) * 9;
        for (int u = 0; u < 3; ++u)
            for (int v = 0; v < 3; ++v)
                s += dk[u * 3 + v] * patch[co][i + u + 5][j + v + 5];
        out16[((size_t)b * 16 + co) * HW + (size_t)h * W + w] = s;
    }
}

// ---------------------------------------------------------------------------
// Window attention: 10x10 windows, 8 heads, ch=8. One block per window,
// qkv channel layout = qi*64 + head*8 + cc. Output: 64-ch pre-projection.
// ---------------------------------------------------------------------------
__global__ void win_attn_k(const float* __restrict__ qkv,
                           const float* __restrict__ rpb,
                           float* __restrict__ out,
                           int H, int W)
{
    __shared__ float qs[100][8], ks[100][8], vs[100][8];
    __shared__ float sc[100][100];

    const int HW = H * W;
    const int b  = blockIdx.z;
    const int wh = blockIdx.y, ww = blockIdx.x;
    const float scale = 0.35355339059327373f;   // 1/sqrt(8)

    for (int hd = 0; hd < 8; ++hd) {
        for (int idx = threadIdx.x; idx < 800; idx += 256) {
            int i = idx >> 3, cc = idx & 7;
            int hg = wh * 10 + i / 10, wg = ww * 10 + i % 10;
            size_t pix = (size_t)hg * W + wg;
            int cq = hd * 8 + cc;
            qs[i][cc] = qkv[((size_t)b * 192 + cq) * HW + pix] * scale;
            ks[i][cc] = qkv[((size_t)b * 192 + 64 + cq) * HW + pix];
            vs[i][cc] = qkv[((size_t)b * 192 + 128 + cq) * HW + pix];
        }
        __syncthreads();

        for (int s = threadIdx.x; s < 10000; s += 256) {
            int i = s / 100, j = s - i * 100;
            float d = 0.0f;
            for (int c = 0; c < 8; ++c) d += qs[i][c] * ks[j][c];
            int ih = i / 10, iw = i % 10, jh = j / 10, jw = j % 10;
            int rel = (jh - ih + 9) * 19 + (jw - iw + 9);
            sc[i][j] = d + rpb[hd * 361 + rel];
        }
        __syncthreads();

        if (threadIdx.x < 100) {
            int i = threadIdx.x;
            float m = -3.0e38f;
            for (int j = 0; j < 100; ++j) m = fmaxf(m, sc[i][j]);
            float sum = 0.0f;
            for (int j = 0; j < 100; ++j) { float e = __expf(sc[i][j] - m); sc[i][j] = e; sum += e; }
            float inv = 1.0f / sum;
            for (int j = 0; j < 100; ++j) sc[i][j] *= inv;
        }
        __syncthreads();

        for (int s = threadIdx.x; s < 800; s += 256) {
            int i = s >> 3, cc = s & 7;
            float acc = 0.0f;
            for (int j = 0; j < 100; ++j) acc += sc[i][j] * vs[j][cc];
            int hg = wh * 10 + i / 10, wg = ww * 10 + i % 10;
            out[((size_t)b * 64 + hd * 8 + cc) * HW + (size_t)hg * W + wg] = acc;
        }
        __syncthreads();
    }
}

// ---------------------------------------------------------------------------
// Final 3x3 conv (64->64, pad 1) fused with bias + outer skip add.
// 8x8 spatial tile per block; 64x10x10 input patch in LDS.
// ---------------------------------------------------------------------------
__global__ void conv3x3_skip_k(const float* __restrict__ in,
                               const float* __restrict__ wgt,
                               const float* __restrict__ bias,
                               const float* __restrict__ skip,
                               float* __restrict__ out,
                               int H, int W)
{
    __shared__ float patch[64][10][10];   // ~25.6 KB
    const int HW = H * W;
    const int b  = blockIdx.z;
    const int h0 = blockIdx.y * 8 - 1;
    const int w0 = blockIdx.x * 8 - 1;

    for (int idx = threadIdx.x; idx < 64 * 100; idx += 256) {
        int ci = idx / 100;
        int r  = idx - ci * 100;
        int i = r / 10, j = r - i * 10;
        int yy = h0 + i, xx = w0 + j;
        float v = 0.0f;
        if (yy >= 0 && yy < H && xx >= 0 && xx < W)
            v = in[((size_t)b * 64 + ci) * HW + (size_t)yy * W + xx];
        patch[ci][i][j] = v;
    }
    __syncthreads();

    int pix = threadIdx.x & 63;
    int i = pix >> 3, j = pix & 7;
    int cog = threadIdx.x >> 6;           // 0..3, 16 output channels each
    int h = blockIdx.y * 8 + i, w = blockIdx.x * 8 + j;

    for (int co = cog * 16; co < cog * 16 + 16; ++co) {
        float s = bias[co];
        for (int ci = 0; ci < 64; ++ci) {
            const float* wp = wgt + ((size_t)co * 64 + ci) * 9;
            float p00 = patch[ci][i][j],     p01 = patch[ci][i][j + 1],     p02 = patch[ci][i][j + 2];
            float p10 = patch[ci][i + 1][j], p11 = patch[ci][i + 1][j + 1], p12 = patch[ci][i + 1][j + 2];
            float p20 = patch[ci][i + 2][j], p21 = patch[ci][i + 2][j + 1], p22 = patch[ci][i + 2][j + 2];
            s += wp[0] * p00 + wp[1] * p01 + wp[2] * p02
               + wp[3] * p10 + wp[4] * p11 + wp[5] * p12
               + wp[6] * p20 + wp[7] * p21 + wp[8] * p22;
        }
        size_t o = ((size_t)b * 64 + co) * HW + (size_t)h * W + w;
        out[o] = s + skip[o];
    }
}

// ---------------------------------------------------------------------------
// Orchestration
// ---------------------------------------------------------------------------
extern "C" void kernel_launch(void* const* d_in, const int* in_sizes, int n_in,
                              void* d_out, int out_size, void* d_ws, size_t ws_size,
                              hipStream_t stream)
{
    (void)in_sizes; (void)n_in; (void)out_size; (void)ws_size;
    const int B = 4, C = 64, H = 200, W = 200, HW = H * W;

    const float* x          = (const float*)d_in[0];
    const float* ln_proj_w  = (const float*)d_in[1];
    const float* ln_proj_b  = (const float*)d_in[2];
    const float* proj_p_w   = (const float*)d_in[3];
    const float* proj_p_b   = (const float*)d_in[4];
    const float* proj_d_w   = (const float*)d_in[5];
    const float* proj_d_b   = (const float*)d_in[6];
    const float* proj_a_w   = (const float*)d_in[7];
    const float* proj_a_b   = (const float*)d_in[8];
    const float* ln_attn_w  = (const float*)d_in[9];
    const float* ln_attn_b  = (const float*)d_in[10];
    const float* qkv_w      = (const float*)d_in[11];
    const float* qkv_b      = (const float*)d_in[12];
    const float* attn_out_w = (const float*)d_in[13];
    const float* attn_out_b = (const float*)d_in[14];
    const float* rpb        = (const float*)d_in[15];
    const float* dwcp1_w    = (const float*)d_in[16];
    const float* dwcp1_b    = (const float*)d_in[17];
    const float* dwcp2_w    = (const float*)d_in[18];
    const float* dwcp2_b    = (const float*)d_in[19];
    const float* pconv_w    = (const float*)d_in[20];
    const float* pconv_b    = (const float*)d_in[21];
    const float* ffn_p_w    = (const float*)d_in[22];
    const float* ffn_p_b    = (const float*)d_in[23];
    const float* ffn_d_w    = (const float*)d_in[24];
    const float* ffn_d_b    = (const float*)d_in[25];
    const float* ffn_a_w    = (const float*)d_in[26];
    const float* ffn_a_b    = (const float*)d_in[27];
    const float* ln_out_w   = (const float*)d_in[28];
    const float* ln_out_b   = (const float*)d_in[29];
    const float* conv_out_w = (const float*)d_in[30];
    const float* conv_out_b = (const float*)d_in[31];
    const float* plk        = (const float*)d_in[32];

    float* out = (float*)d_out;

    // scratch layout (floats)
    float* ws = (float*)d_ws;
    const size_t nBHW = (size_t)B * HW;          // 160,000
    float* bufA = ws;                            // 192 ch (qkv / dw outputs)
    float* bufB = bufA + 192 * nBHW;             // 128 ch (hidden)
    float* bufC = bufB + 128 * nBHW;             // 64 ch  (running x)
    float* bufD = bufC + 64 * nBHW;              // 64 ch
    float* bufE = bufD + 64 * nBHW;              // 16 ch  (conv_attention out)
    float* lk   = bufE + 16 * nBHW;              // 16*16*169
    float* gbuf = lk + 16 * 16 * 169;            // B*16
    float* dynk = gbuf + B * 16;                 // B*144

    dim3 blk(256);
    auto cdiv = [](int n) { return (n + 255) / 256; };
    const int gx = (HW + 63) / 64;               // 625
    auto gemm_grid = [&](int Cout) { return dim3(gx, (Cout + 63) / 64, B); };

    // ---- x = layernorm(x, ln_proj); conv_ffn(proj) ----
    layernorm_cf_k<<<cdiv(B * HW), blk, 0, stream>>>(x, ln_proj_w, ln_proj_b, bufC, C, HW, B * HW);
    gemm1x1_wmma_k<<<gemm_grid(128), blk, 0, stream>>>(bufC, proj_p_w, proj_p_b, bufB, 64, 128, HW, 1);
    dw3x3_gelu_add_k<<<cdiv(B * 128 * HW), blk, 0, stream>>>(bufB, proj_d_w, proj_d_b, bufA, 128, H, W, B * 128 * HW);
    gemm1x1_wmma_k<<<gemm_grid(64), blk, 0, stream>>>(bufA, proj_a_w, proj_a_b, bufC, 128, 64, HW, 0);

    // ---- x = x + window_attention(LN(x)) ----
    layernorm_cf_k<<<cdiv(B * HW), blk, 0, stream>>>(bufC, ln_attn_w, ln_attn_b, bufD, C, HW, B * HW);
    gemm1x1_wmma_k<<<gemm_grid(192), blk, 0, stream>>>(bufD, qkv_w, qkv_b, bufA, 64, 192, HW, 0);
    win_attn_k<<<dim3(20, 20, B), blk, 0, stream>>>(bufA, rpb, bufD, H, W);
    gemm1x1_wmma_k<<<gemm_grid(64), blk, 0, stream>>>(bufD, attn_out_w, attn_out_b, bufB, 64, 64, HW, 0);
    add_inplace_k<<<cdiv((int)(64 * nBHW)), blk, 0, stream>>>(bufC, bufB, 64 * (int)nBHW);

    // ---- lk = geo_ensemble(plk) ----
    geo_ensemble_k<<<cdiv(16 * 16 * 169), blk, 0, stream>>>(plk, lk, 16 * 16 * 169);

    // ---- NBLK = 2 residual blocks ----
    for (int i = 0; i < 2; ++i) {
        gemm1x1_wmma_k<<<gemm_grid(80), blk, 0, stream>>>(
            bufC, ffn_p_w + i * 80 * 64, ffn_p_b + i * 80, bufB, 64, 80, HW, 1);
        dw3x3_gelu_add_k<<<cdiv(B * 80 * HW), blk, 0, stream>>>(
            bufB, ffn_d_w + i * 80 * 9, ffn_d_b + i * 80, bufA, 80, H, W, B * 80 * HW);
        gemm1x1_wmma_k<<<gemm_grid(64), blk, 0, stream>>>(
            bufA, ffn_a_w + i * 64 * 80, ffn_a_b + i * 64, bufD, 80, 64, HW, 0);

        // conv_attention on y (= bufD)
        gmean16_k<<<B * 16, blk, 0, stream>>>(bufD, gbuf, HW);
        dyn_kernels_k<<<B, blk, 0, stream>>>(gbuf,
            dwcp1_w + i * 8 * 16, dwcp1_b + i * 8,
            dwcp2_w + i * 144 * 8, dwcp2_b + i * 144, dynk);
        conv_attn16_k<<<dim3((W + 15) / 16, (H + 15) / 16, B), blk, 0, stream>>>(
            bufD, lk, dynk, bufE, H, W);
        copy16_k<<<cdiv((int)(16 * nBHW)), blk, 0, stream>>>(bufE, bufD, HW, 16 * (int)nBHW);

        gemm1x1_wmma_k<<<gemm_grid(64), blk, 0, stream>>>(
            bufD, pconv_w + i * 64 * 64, pconv_b + i * 64, bufB, 64, 64, HW, 0);
        add_inplace_k<<<cdiv((int)(64 * nBHW)), blk, 0, stream>>>(bufC, bufB, 64 * (int)nBHW);
    }

    // ---- out = conv3x3(LN(x)) + skip ----
    layernorm_cf_k<<<cdiv(B * HW), blk, 0, stream>>>(bufC, ln_out_w, ln_out_b, bufD, C, HW, B * HW);
    conv3x3_skip_k<<<dim3(W / 8, H / 8, B), blk, 0, stream>>>(
        bufD, conv_out_w, conv_out_b, x, out, H, W);
}
